// loss_function_new_79070347919370
// MI455X (gfx1250) — compile-verified, matrix operand order checked by reading, not run on hardware
//
#include <hip/hip_runtime.h>
#include <hip/hip_bf16.h>

typedef __attribute__((ext_vector_type(16))) _Float16 v16h;
typedef __attribute__((ext_vector_type(2)))  _Float16 v2h;
typedef __attribute__((ext_vector_type(8)))  float    v8f;
typedef __attribute__((ext_vector_type(4)))  int      v4i;

#define BATCH 1024
#define N_ENT 40000
#define DIM   128
#define TM    64      // rows (batch) per block
#define TN    64      // cols (entities) per block
#define LOG_EPS   (-16.11809565f)      // logf(1e-7)
#define LOG_1MEPS (-1.0000001e-7f)     // logf(1 - 1e-7)

// ---------------- workspace byte layout ----------------
#define WS_ACC    0
#define WS_CPOS   64
#define WS_CNEG   (WS_CPOS + BATCH*DIM*2)
#define WS_NPOS   (WS_CNEG + BATCH*DIM*2)
#define WS_NNEG   (WS_NPOS + BATCH*4)
#define WS_ENORM  (WS_NNEG + BATCH*4)

#if defined(__has_builtin)
#if __has_builtin(__builtin_amdgcn_global_load_async_to_lds_b128)
#define HAVE_ASYNC_LDS 1
#endif
#if __has_builtin(__builtin_amdgcn_s_wait_asynccnt)
#define HAVE_WAIT_ASYNC 1
#endif
#endif

__global__ void init_acc_kernel(float* acc) {
    if (threadIdx.x < 4) acc[threadIdx.x] = 0.0f;
}

// c[i] = ent[h[i]] + rseq[i]  -> f16 copy + fp32 squared norm. grid=B, block=128
__global__ void prep_c_kernel(const float* __restrict__ ent,
                              const int*   __restrict__ h_idx,
                              const float* __restrict__ rseq,
                              _Float16* __restrict__ c_h,
                              float* __restrict__ cnorm) {
    int i = blockIdx.x;
    int k = threadIdx.x;
    float v = ent[(size_t)h_idx[i] * DIM + k] + rseq[(size_t)i * DIM + k];
    c_h[(size_t)i * DIM + k] = (_Float16)v;
    float s = v * v;
    #pragma unroll
    for (int o = 16; o > 0; o >>= 1) s += __shfl_down(s, o, 32);
    __shared__ float red[4];
    if ((threadIdx.x & 31) == 0) red[threadIdx.x >> 5] = s;
    __syncthreads();
    if (threadIdx.x == 0) cnorm[i] = red[0] + red[1] + red[2] + red[3];
}

// enorm[j] = ||ent[j]||^2, one wave per row. grid = N/8, block = 256 (8 waves)
__global__ void enorm_kernel(const float* __restrict__ ent, float* __restrict__ enorm) {
    int wave = threadIdx.x >> 5;
    int lane = threadIdx.x & 31;
    int row  = blockIdx.x * 8 + wave;
    const float4 f = *(const float4*)(ent + (size_t)row * DIM + lane * 4);
    float s = f.x * f.x + f.y * f.y + f.z * f.z + f.w * f.w;
    #pragma unroll
    for (int o = 16; o > 0; o >>= 1) s += __shfl_down(s, o, 32);
    if (lane == 0) enorm[row] = s;
}

// Main fused kernel: 64x64 distance tile per block; 8 waves as 2(m-pair)x4(n);
// each wave computes TWO 16x16 WMMA tiles sharing B fragments (8 WMMAs/wave),
// fused rsq/exp/log BCE epilogue reduced to one atomic per block.
__launch_bounds__(256)
__global__ void dist_loss_kernel(const float*    __restrict__ ent,
                                 const _Float16* __restrict__ c_h,
                                 const float*    __restrict__ cnorm,
                                 const float*    __restrict__ enorm,
                                 const int*      __restrict__ t_idx,
                                 const int*      __restrict__ l1_flag,
                                 float* acc_neg, float* acc_corr, int positive) {
    __shared__ __align__(16) _Float16 Ash[TM][DIM + 8];
    __shared__ __align__(16) _Float16 Bsh[TN][DIM + 8];
    __shared__ float cn_sh[TM];
    __shared__ int   tsh[TM];
    __shared__ float rn[8], rc[8];

    const int tid = threadIdx.x;
    const bool l1 = (l1_flag[0] != 0);

    // --- stage A tile (f16 c rows): 64 rows x 128 halfs, 4x b128 per thread ---
    {
        int m  = tid >> 2;
        int k0 = (tid & 3) * 32;
        int gm = blockIdx.y * TM + m;
        const _Float16* src = c_h + (size_t)gm * DIM + k0;
        _Float16* dst = &Ash[m][k0];
#ifdef HAVE_ASYNC_LDS
        #pragma unroll
        for (int q = 0; q < 4; ++q)
            __builtin_amdgcn_global_load_async_to_lds_b128(
                (__attribute__((address_space(1))) v4i*)(src + q * 8),
                (__attribute__((address_space(3))) v4i*)(dst + q * 8),
                0, 0);
#else
        const uint4* s4 = (const uint4*)src;
        uint4* d4 = (uint4*)dst;
        #pragma unroll
        for (int q = 0; q < 4; ++q) d4[q] = s4[q];
#endif
    }
    // --- stage B tile (ent f32 -> f16): 64 rows x 128, 32 floats per thread ---
    {
        int j  = tid >> 2;
        int k0 = (tid & 3) * 32;
        int gj = blockIdx.x * TN + j;
        const float4* src = (const float4*)(ent + (size_t)gj * DIM + k0);
        #pragma unroll
        for (int q = 0; q < 8; ++q) {
            float4 f = src[q];
            v2h p0; p0.x = (_Float16)f.x; p0.y = (_Float16)f.y;
            v2h p1; p1.x = (_Float16)f.z; p1.y = (_Float16)f.w;
            *(v2h*)&Bsh[j][k0 + q * 4]     = p0;
            *(v2h*)&Bsh[j][k0 + q * 4 + 2] = p1;
        }
    }
    if (tid < TM) {
        cn_sh[tid] = cnorm[blockIdx.y * TM + tid];
        tsh[tid]   = positive ? t_idx[blockIdx.y * TM + tid] : -1;
    }
#ifdef HAVE_ASYNC_LDS
#ifdef HAVE_WAIT_ASYNC
    __builtin_amdgcn_s_wait_asynccnt(0);
#else
    asm volatile("s_wait_asynccnt 0" ::: "memory");
#endif
#endif
    __syncthreads();

    const int wave = tid >> 5;
    const int lane = tid & 31;
    const int l    = lane & 15;
    const int h    = lane >> 4;
    const int n0   = (wave & 3) * 16;       // n-subtile 0..48
    const int mp   = wave >> 2;             // m-pair 0/1 -> tiles mp, mp+2
    const int m0a  = mp * 16;
    const int m0b  = (mp + 2) * 16;
    const int gj   = blockIdx.x * TN + n0 + l;

    float sarr[16];   // logit s = 100/dist for 16 output elements per lane

    if (!l1) {
        v8f acc0 = {}, acc1 = {};
        #pragma unroll
        for (int kk = 0; kk < 4; ++kk) {
            v16h a0, a1, b;
            const int kb = kk * 32;
            #pragma unroll
            for (int v = 0; v < 8; ++v) {
                int k = kb + (v & 3) * 2 + (v >> 2) * 16 + h * 8;
                v2h pb = *(const v2h*)&Bsh[n0 + l][k];
                v2h p0 = *(const v2h*)&Ash[m0a + l][k];
                v2h p1 = *(const v2h*)&Ash[m0b + l][k];
                b[2 * v] = pb.x;  b[2 * v + 1] = pb.y;
                a0[2 * v] = p0.x; a0[2 * v + 1] = p0.y;
                a1[2 * v] = p1.x; a1[2 * v + 1] = p1.y;
            }
            acc0 = __builtin_amdgcn_wmma_f32_16x16x32_f16(
                       false, a0, false, b, (short)0, acc0, false, false);
            acc1 = __builtin_amdgcn_wmma_f32_16x16x32_f16(
                       false, a1, false, b, (short)0, acc1, false, false);
        }
        const float en = enorm[gj];
        #pragma unroll
        for (int r = 0; r < 8; ++r) {
            float sq0 = cn_sh[m0a + r + 8 * h] + en - 2.0f * acc0[r];
            float sq1 = cn_sh[m0b + r + 8 * h] + en - 2.0f * acc1[r];
            // s = 100/dist = 100 * rsqrt(sq); raw v_rsq_f32, clamp handles sq->0
            sarr[r]     = 100.0f * __builtin_amdgcn_rsqf(fmaxf(sq0, 1e-24f));
            sarr[8 + r] = 100.0f * __builtin_amdgcn_rsqf(fmaxf(sq1, 1e-24f));
        }
    } else {
        // ---- L1 (Manhattan) fallback: direct VALU reduction from LDS ----
        #pragma unroll
        for (int e = 0; e < 16; ++e) {
            int m = (e < 8 ? m0a : m0b) + (e & 7) + 8 * h;
            float d = 0.0f;
            for (int k = 0; k < DIM; ++k)
                d += fabsf((float)Ash[m][k] - (float)Bsh[n0 + l][k]);
            sarr[e] = 100.0f * __builtin_amdgcn_rcpf(fmaxf(d, 1e-12f));
        }
    }

    // ---- branchless fused BCE epilogue ----
    float local_neg = 0.0f, local_corr = 0.0f;
    #pragma unroll
    for (int e = 0; e < 16; ++e) {
        int m = (e < 8 ? m0a : m0b) + (e & 7) + 8 * h;
        float s   = sarr[e];
        float t   = __expf(-s);
        float lse = __logf(1.0f + t);                 // softplus(-s), t in (0,1]
        float lneg = fmaxf(-s - lse, LOG_EPS);        // log(max(1-pred, eps))
        local_neg += lneg;
        float lpos = fminf(-lse, LOG_1MEPS);          // log(min(pred, 1-eps))
        local_corr += (tsh[m] == gj) ? (lpos - lneg) : 0.0f;
    }

    // ---- wave + block reduction, one atomic per block ----
    #pragma unroll
    for (int o = 16; o > 0; o >>= 1) {
        local_neg  += __shfl_down(local_neg,  o, 32);
        local_corr += __shfl_down(local_corr, o, 32);
    }
    if (lane == 0) { rn[wave] = local_neg; rc[wave] = local_corr; }
    __syncthreads();
    if (tid == 0) {
        float sn = 0.0f, sc = 0.0f;
        #pragma unroll
        for (int w = 0; w < 8; ++w) { sn += rn[w]; sc += rc[w]; }
        atomicAdd(acc_neg, sn);
        if (positive) atomicAdd(acc_corr, sc);
    }
}

__global__ void finalize_kernel(const float* acc, float* out) {
    const float invBN = 1.0f / ((float)BATCH * (float)N_ENT);
    out[0] = -(acc[0] + acc[1]) * invBN - acc[2] * invBN;
}

extern "C" void kernel_launch(void* const* d_in, const int* in_sizes, int n_in,
                              void* d_out, int out_size, void* d_ws, size_t ws_size,
                              hipStream_t stream) {
    const int*   pos_h = (const int*)  d_in[0];
    const int*   pos_t = (const int*)  d_in[1];
    const int*   neg_h = (const int*)  d_in[2];
    const float* rs_p  = (const float*)d_in[4];
    const float* rs_n  = (const float*)d_in[5];
    const float* ent   = (const float*)d_in[6];
    const int*   l1f   = (const int*)  d_in[7];

    char* ws = (char*)d_ws;
    float*    acc      = (float*)   (ws + WS_ACC);
    _Float16* c_pos    = (_Float16*)(ws + WS_CPOS);
    _Float16* c_neg    = (_Float16*)(ws + WS_CNEG);
    float*    cn_pos   = (float*)   (ws + WS_NPOS);
    float*    cn_neg   = (float*)   (ws + WS_NNEG);
    float*    enorm    = (float*)   (ws + WS_ENORM);
    float*    out      = (float*)   d_out;

    init_acc_kernel<<<1, 64, 0, stream>>>(acc);
    prep_c_kernel<<<BATCH, 128, 0, stream>>>(ent, pos_h, rs_p, c_pos, cn_pos);
    prep_c_kernel<<<BATCH, 128, 0, stream>>>(ent, neg_h, rs_n, c_neg, cn_neg);
    enorm_kernel<<<N_ENT / 8, 256, 0, stream>>>(ent, enorm);

    dim3 grid(N_ENT / TN, BATCH / TM);   // 625 x 16
    dist_loss_kernel<<<grid, 256, 0, stream>>>(ent, c_pos, cn_pos, enorm, pos_t, l1f,
                                               acc + 0, acc + 1, 1);
    dist_loss_kernel<<<grid, 256, 0, stream>>>(ent, c_neg, cn_neg, enorm, pos_t, l1f,
                                               acc + 2, acc + 3, 0);
    finalize_kernel<<<1, 1, 0, stream>>>(acc, out);
}
